// Breadth_15917148799860
// MI455X (gfx1250) — matrix operand order repaired and checked by reading, not compiled, not run on hardware
//
#include <hip/hip_runtime.h>
#include <hip/hip_bf16.h>

typedef __attribute__((ext_vector_type(16))) _Float16 v16h;
typedef __attribute__((ext_vector_type(8)))  float    v8f;
typedef __attribute__((ext_vector_type(2)))  _Float16 h2v;
typedef __attribute__((ext_vector_type(2)))  float    f2v;

#define KDIM   256
#define HC     128
#define HEADS  4
#define CH     32
#define NEG_SLOPE 0.2f
#define WT_STRIDE 132   // padded dword stride: 4*n mod 64 tiles all LDS banks

// ---------------------------------------------------------------------------
// Kernel 0: zero the output accumulator, set m = -inf, s = 0
// ---------------------------------------------------------------------------
__global__ void init_kernel(float* __restrict__ out, float* __restrict__ m,
                            float* __restrict__ s, int total_out, int nh) {
    int idx = blockIdx.x * blockDim.x + threadIdx.x;
    if (idx < total_out) out[idx] = 0.0f;
    if (idx < nh) {
        m[idx] = -__builtin_inff();
        s[idx] = 0.0f;
    }
}

// ---------------------------------------------------------------------------
// Kernel 1: h = x @ W via v_wmma_f32_16x16x32_f16.
// Block = 256 threads = 8 waves. W staged in LDS transposed + pre-packed as
// f16x2 per dword: WT[n][kp] = (W[2kp][n], W[2kp+1][n]). A lane's B fragment
// is two 4-dword runs 8 dwords apart -> two ds_load_b128, no packing movs.
// Each wave computes a 16-row x 128-col strip: 8 C tiles, 8 k-steps of K=32.
// ---------------------------------------------------------------------------
__global__ __launch_bounds__(256) void gemm_wmma_kernel(
        const float* __restrict__ x, const float* __restrict__ W,
        float* __restrict__ h, int nrows) {
    __shared__ unsigned int WT[HC * WT_STRIDE];   // 128*132*4 = 67.6 KB

    // Cooperative stage: consecutive threads -> consecutive n (coalesced W reads)
    for (int idx = threadIdx.x; idx < HC * (KDIM / 2); idx += 256) {
        const int n  = idx & (HC - 1);
        const int kp = idx >> 7;
        const _Float16 lo = (_Float16)W[(2 * kp) * HC + n];
        const _Float16 hi = (_Float16)W[(2 * kp + 1) * HC + n];
        const unsigned int p =
            (unsigned int)__builtin_bit_cast(unsigned short, lo) |
            ((unsigned int)__builtin_bit_cast(unsigned short, hi) << 16);
        WT[n * WT_STRIDE + kp] = p;
    }
    __syncthreads();

    const int wave    = threadIdx.x >> 5;
    const int lane    = threadIdx.x & 31;
    const int strip   = blockIdx.x * 8 + wave;
    const int rowbase = strip * 16;
    if (rowbase >= nrows) return;

    const int mrow = lane & 15;   // A M index / B N index / C N index
    const int half = lane >> 4;   // selects K sub-range

    const float* __restrict__ xrow = x + (size_t)(rowbase + mrow) * KDIM;

    v8f acc[8] = {};   // 8 output tiles of 16x16 f32

    for (int ks = 0; ks < 8; ++ks) {
        const int kbase = ks * 32;
        __builtin_prefetch(xrow + kbase + 32, 0, 3);   // speculative: safe OOB

        // A fragment (16x32 f16). ISA layout:
        //   VGPR j holds k = (j>>2)*16 + half*8 + (j&3)*2 (+1)
        union { v16h v; h2v p[8]; } au;
#pragma unroll
        for (int j = 0; j < 8; ++j) {
            const int k0 = kbase + ((j >> 2) << 4) + (half << 3) + ((j & 3) << 1);
            const f2v xf = *(const f2v*)(xrow + k0);
            au.p[j] = __builtin_convertvector(xf, h2v);
        }

        const int kp0 = ks * 16 + (half << 2);
#pragma unroll
        for (int t = 0; t < 8; ++t) {
            const int ncol = t * 16 + mrow;
            const unsigned int* __restrict__ bp = &WT[ncol * WT_STRIDE + kp0];
            union { v16h v; int4 q[2]; } bu;
            bu.q[0] = *(const int4*)(bp);       // ds_load_b128
            bu.q[1] = *(const int4*)(bp + 8);   // ds_load_b128
            acc[t] = __builtin_amdgcn_wmma_f32_16x16x32_f16(
                false, au.v, false, bu.v, (short)0, acc[t], false, false);
        }
    }

    // Store C tiles. Layout: VGPR r, lane l -> row = r + (l>>4)*8, col = l&15
#pragma unroll
    for (int t = 0; t < 8; ++t) {
#pragma unroll
        for (int r = 0; r < 8; ++r) {
            const int row = rowbase + r + (half << 3);
            const int col = t * 16 + mrow;
            h[(size_t)row * HC + col] = acc[t][r];
        }
    }
}

// ---------------------------------------------------------------------------
// Kernel 2: per-(node, head) attention logits alpha_s / alpha_d
// ---------------------------------------------------------------------------
__global__ void alpha_kernel(const float* __restrict__ h,
                             const float* __restrict__ a_src,
                             const float* __restrict__ a_dst,
                             float* __restrict__ as, float* __restrict__ ad,
                             int n) {
    int idx = blockIdx.x * blockDim.x + threadIdx.x;
    if (idx >= n * HEADS) return;
    const int node = idx >> 2;
    const int head = idx & 3;
    const float* __restrict__ hp = h + (size_t)node * HC + head * CH;
    const float* __restrict__ vs = a_src + head * CH;
    const float* __restrict__ vd = a_dst + head * CH;
    float ss = 0.0f, dd = 0.0f;
#pragma unroll
    for (int c = 0; c < CH; ++c) {
        const float v = hp[c];
        ss += v * vs[c];
        dd += v * vd[c];
    }
    as[idx] = ss;
    ad[idx] = dd;
}

// ---------------------------------------------------------------------------
// Float atomic-max via int-max / uint-min trick (init must be -inf)
// ---------------------------------------------------------------------------
__device__ __forceinline__ void atomicMaxFloat(float* addr, float val) {
    if (val >= 0.0f)
        atomicMax((int*)addr, __float_as_int(val));
    else
        atomicMin((unsigned int*)addr, __float_as_uint(val));
}

__device__ __forceinline__ float leaky(float v) {
    return v > 0.0f ? v : v * NEG_SLOPE;
}

// ---------------------------------------------------------------------------
// Kernel 3: segment-max of edge logits into m[dst, head]
// ---------------------------------------------------------------------------
__global__ void edge_max_kernel(const int* __restrict__ ei,
                                const float* __restrict__ as,
                                const float* __restrict__ ad,
                                float* __restrict__ m, int E_, int n) {
    int e = blockIdx.x * blockDim.x + threadIdx.x;
    if (e >= E_ + n) return;
    int src, dst;
    if (e < E_) { src = ei[e]; dst = ei[E_ + e]; }
    else        { src = dst = e - E_; }
#pragma unroll
    for (int hh = 0; hh < HEADS; ++hh) {
        const float v = leaky(as[src * HEADS + hh] + ad[dst * HEADS + hh]);
        atomicMaxFloat(&m[dst * HEADS + hh], v);
    }
}

// ---------------------------------------------------------------------------
// Kernel 4: one wave32 per edge. Accumulate numerator (p * h[src]) into out
// and p into s. Normalization deferred to finalize (out/s is exact fusion).
// ---------------------------------------------------------------------------
__global__ __launch_bounds__(256) void edge_scatter_kernel(
        const int* __restrict__ ei, const float* __restrict__ as,
        const float* __restrict__ ad, const float* __restrict__ m,
        const float* __restrict__ h, float* __restrict__ s,
        float* __restrict__ out, int E_, int n) {
    const int gid  = blockIdx.x * blockDim.x + threadIdx.x;
    const int e    = gid >> 5;
    const int lane = gid & 31;
    if (e >= E_ + n) return;
    int src, dst;
    if (e < E_) { src = ei[e]; dst = ei[E_ + e]; }
    else        { src = dst = e - E_; }

    float p[HEADS];
#pragma unroll
    for (int hh = 0; hh < HEADS; ++hh) {
        const float v = leaky(as[src * HEADS + hh] + ad[dst * HEADS + hh]);
        p[hh] = __expf(v - m[dst * HEADS + hh]);
    }
    if (lane < HEADS) atomicAdd(&s[dst * HEADS + lane], p[lane]);

    const float* __restrict__ hs = h + (size_t)src * HC;
    float* __restrict__ od = out + (size_t)dst * HC;
#pragma unroll
    for (int hh = 0; hh < HEADS; ++hh)
        atomicAdd(&od[hh * CH + lane], p[hh] * hs[hh * CH + lane]);
}

// ---------------------------------------------------------------------------
// Kernel 5: out = tanh(out / (s + eps) + bias)
// ---------------------------------------------------------------------------
__global__ void finalize_kernel(float* __restrict__ out,
                                const float* __restrict__ s,
                                const float* __restrict__ bias, int n) {
    int idx = blockIdx.x * blockDim.x + threadIdx.x;
    if (idx >= n * HC) return;
    const int node = idx >> 7;
    const int f    = idx & 127;
    const int hh   = f >> 5;
    const float denom = s[node * HEADS + hh] + 1e-16f;
    out[idx] = tanhf(out[idx] / denom + bias[f]);
}

// ---------------------------------------------------------------------------
extern "C" void kernel_launch(void* const* d_in, const int* in_sizes, int n_in,
                              void* d_out, int out_size, void* d_ws, size_t ws_size,
                              hipStream_t stream) {
    const float* x     = (const float*)d_in[0];
    const float* W     = (const float*)d_in[1];
    const float* a_src = (const float*)d_in[2];
    const float* a_dst = (const float*)d_in[3];
    const float* bias  = (const float*)d_in[4];
    const int*   ei    = (const int*)d_in[5];

    const int n  = in_sizes[0] / KDIM;   // 100000
    const int E_ = in_sizes[5] / 2;      // 1000000

    float* out = (float*)d_out;
    float* ws  = (float*)d_ws;
    float* h   = ws;                                    // n*128 floats
    float* as  = h  + (size_t)n * HC;                   // n*4
    float* ad  = as + (size_t)n * HEADS;                // n*4
    float* m   = ad + (size_t)n * HEADS;                // n*4
    float* s   = m  + (size_t)n * HEADS;                // n*4

    const int total_out = n * HC;
    init_kernel<<<(total_out + 255) / 256, 256, 0, stream>>>(out, m, s,
                                                             total_out, n * HEADS);

    const int nstrips = (n + 15) / 16;
    gemm_wmma_kernel<<<(nstrips + 7) / 8, 256, 0, stream>>>(x, W, h, n);

    alpha_kernel<<<(n * HEADS + 255) / 256, 256, 0, stream>>>(h, a_src, a_dst,
                                                              as, ad, n);

    const int total_e = E_ + n;
    edge_max_kernel<<<(total_e + 255) / 256, 256, 0, stream>>>(ei, as, ad, m,
                                                               E_, n);

    const long long sthreads = (long long)total_e * 32;
    edge_scatter_kernel<<<(int)((sthreads + 255) / 256), 256, 0, stream>>>(
        ei, as, ad, m, h, s, out, E_, n);

    finalize_kernel<<<(total_out + 255) / 256, 256, 0, stream>>>(out, s, bias, n);
}
